// LenaTrans_23484881174857
// MI455X (gfx1250) — matrix operand order repaired and checked by reading, not compiled, not run on hardware
//
#include <hip/hip_runtime.h>
#include <hip/hip_bf16.h>
#include <cstdint>

#define B_    1024
#define S_    139
#define NTOK  (B_*S_)        // 142336, divisible by 128
#define H_    128
#define LDA   130            // padded LDS stride (bf16 elems)
#define WIMG_ELEMS (128*LDA)          // 16640 bf16 = 33280 bytes per weight image
#define WIMG_BYTES (WIMG_ELEMS*2)
#define INV_SCALE 0.17677669529663687f   // 1/sqrt(32)
#define NEGBIG (-1.0e9f)

typedef __attribute__((ext_vector_type(16))) __bf16 v16bf;
typedef __attribute__((ext_vector_type(8)))  float  v8f;
typedef __attribute__((ext_vector_type(4)))  unsigned int v4u;
typedef __attribute__((ext_vector_type(8)))  int v8i;
typedef __attribute__((ext_vector_type(4)))  int v4i;

// ---------------- wave helpers (wave32) ----------------
__device__ inline float wredsum(float v){
  #pragma unroll
  for (int i = 16; i > 0; i >>= 1) v += __shfl_xor(v, i);
  return v;
}
__device__ inline int iredsum(int v){
  #pragma unroll
  for (int i = 16; i > 0; i >>= 1) v += __shfl_xor(v, i);
  return v;
}

// ---------------- TDM: async DMA of one 33280B weight image global->LDS ----
// 1-D copy as 4160 x 8-byte elements. Group0: count=1, type=2 (image).
// Group1: data_size=3 (8B), tensor_dim0=tile_dim0=4160, tensor_dim1=tile_dim1=1.
__device__ inline void tdm_load_w(const void* gsrc, unsigned int lds_byte_off){
  const uint64_t ga = (uint64_t)(uintptr_t)gsrc;
  v4u g0;
  g0[0] = 1u;                                   // count=1, user mode
  g0[1] = lds_byte_off;                         // lds_addr
  g0[2] = (unsigned int)ga;                     // global_addr[31:0]
  g0[3] = (unsigned int)(ga >> 32) | 0x80000000u; // global_addr[56:32] | type=2<<30
  v8i g1;
  g1[0] = (int)(3u << 16);                      // workgroup_mask=0 | data_size=3
  g1[1] = (int)(4160u << 16);                   // tensor_dim0[15:0]
  g1[2] = (int)(1u << 16);                      // tensor_dim0[31:16]=0 | tensor_dim1 lo=1
  g1[3] = (int)(4160u << 16);                   // tensor_dim1 hi=0 | tile_dim0=4160
  g1[4] = 1;                                    // tile_dim1=1 | tile_dim2=0
  g1[5] = 4160;                                 // tensor_dim0_stride[31:0]
  g1[6] = (int)(4160u << 16);                   // stride0 hi=0 | stride1[15:0]
  g1[7] = 0;                                    // stride1 hi
  v4i z4 = {0, 0, 0, 0};
#if __clang_major__ >= 23
  v8i z8 = {0, 0, 0, 0, 0, 0, 0, 0};
  __builtin_amdgcn_tensor_load_to_lds(g0, g1, z4, z4, z8, 0);
#else
  __builtin_amdgcn_tensor_load_to_lds(g0, g1, z4, z4, 0);
#endif
}

// ---------------- WMMA fragment loaders ----------------
// A: 16x32 bf16, rows m striped per ISA layout (lanes 0-15: M, K groups by lane>>4)
__device__ inline v16bf load_a_frag(const __bf16* tile, int m0, int k0, int lane){
  const int m = m0 + (lane & 15);
  const int g = lane >> 4;
  v16bf a;
  #pragma unroll
  for (int v = 0; v < 8; ++v){
    const int k = k0 + ((v & 3) << 1) + ((v >> 2) << 4) + (g << 3);
    const __bf16* p = tile + m * LDA + k;
    a[2*v]   = p[0];
    a[2*v+1] = p[1];
  }
  return a;
}
// B: 32x16 bf16 from W^T staged as [n][k] (stride LDA)
__device__ inline v16bf load_b_frag(const __bf16* wt, int n0, int k0, int lane){
  const int n = n0 + (lane & 15);
  const int g = lane >> 4;
  v16bf b;
  #pragma unroll
  for (int v = 0; v < 8; ++v){
    const int k = k0 + ((v & 3) << 1) + ((v >> 2) << 4) + (g << 3);
    const __bf16* p = wt + n * LDA + k;
    b[2*v]   = p[0];
    b[2*v+1] = p[1];
  }
  return b;
}

__device__ inline void zero_acc(v8f acc[8]){
  #pragma unroll
  for (int nt = 0; nt < 8; ++nt){
    #pragma unroll
    for (int j = 0; j < 8; ++j) acc[nt][j] = 0.0f;
  }
}

// 16x128 strip, K=128: 4 k-steps x 8 n-tiles = 32 WMMAs
__device__ inline void gemm_strip(const __bf16* At, const __bf16* Wt, int m0, int lane, v8f acc[8]){
  #pragma unroll
  for (int kt = 0; kt < 4; ++kt){
    v16bf a = load_a_frag(At, m0, kt*32, lane);
    #pragma unroll
    for (int nt = 0; nt < 8; ++nt){
      v16bf b = load_b_frag(Wt, nt*16, kt*32, lane);
      acc[nt] = __builtin_amdgcn_wmma_f32_16x16x32_bf16(false, a, false, b,
                                                        (short)0, acc[nt], false, false);
    }
  }
}

// =====================================================================
// Kernel 0: one-time weight prep. f32 [k][n] -> bf16 W^T image [n][LDA]
// images: 0=map_w, 1=wk, 2=wv, 3=wq
// =====================================================================
__global__ __launch_bounds__(256)
void k_prep_w(const float* __restrict__ map_w, const float* __restrict__ wk,
              const float* __restrict__ wv, const float* __restrict__ wq,
              __bf16* __restrict__ imgs)
{
  const float* W = (blockIdx.x == 0) ? map_w :
                   (blockIdx.x == 1) ? wk :
                   (blockIdx.x == 2) ? wv : wq;
  __bf16* img = imgs + (size_t)blockIdx.x * WIMG_ELEMS;
  for (int idx = threadIdx.x; idx < 128*128; idx += 256){
    const int k = idx >> 7, n = idx & 127;
    img[n*LDA + k] = (__bf16)W[idx];
  }
  for (int n = threadIdx.x; n < 128; n += 256){   // deterministic pad
    img[n*LDA + 128] = (__bf16)0.0f;
    img[n*LDA + 129] = (__bf16)0.0f;
  }
}

// =====================================================================
// Kernel 1: fused embedding gather + LN + (emb@map_w) + (hs@wk) + (hs@wv)
// 128 tokens per block, 256 threads (8 waves). Weights arrive via TDM,
// double-buffered in LDS and overlapped with gather/LN and WMMA.
// LDS: At @0 (33280B), W0 @33280, W1 @66560 -> 99840B dynamic.
// =====================================================================
__global__ __launch_bounds__(256)
void k_embed_map_kv(const int* __restrict__ x,
                    const float* __restrict__ cat_emb, const float* __restrict__ tok_type,
                    const float* __restrict__ lng, const float* __restrict__ lnb,
                    const __bf16* __restrict__ wimg,   // images: map, wk, wv
                    const float* __restrict__ map_b,
                    const float* __restrict__ bk, const float* __restrict__ bv,
                    __bf16* __restrict__ kout, __bf16* __restrict__ vout,
                    float* __restrict__ hs_last, float* __restrict__ maskNeg)
{
  extern __shared__ __bf16 smem[];
  __bf16* At = smem;                    // 128 x LDA activations (emb, later hs)
  __bf16* W0 = smem + WIMG_ELEMS;       // weight buffer 0
  __bf16* W1 = smem + 2*WIMG_ELEMS;     // weight buffer 1

  const int tid  = threadIdx.x;
  const int lane = tid & 31;
  const int w    = tid >> 5;
  const int t0   = blockIdx.x * 128;
  const int m0   = w * 16;
  const int g    = lane >> 4;
  const int nlo  = lane & 15;

  // kick off async weight DMAs before doing any VALU work
  if (w == 0){
    tdm_load_w(wimg,                  (unsigned int)WIMG_BYTES);      // map_w -> W0
    tdm_load_w(wimg + WIMG_ELEMS,     (unsigned int)(2*WIMG_BYTES));  // wk    -> W1
  }

  // ---- Stage A: embedding gather + tok_type + LayerNorm -> At (bf16) ----
  for (int i = 0; i < 16; ++i){
    const int local = m0 + i;
    const int t = t0 + local;
    const int* xr = x + (size_t)t * 38;
    const int c  = lane >> 2;                       // cat group for lanes < 24
    const int id = (lane < 24) ? xr[c] : 0;
    float vals[4];
    #pragma unroll
    for (int j = 0; j < 4; ++j){
      const int e = lane*4 + j;
      float vv;
      if (lane < 24) vv = cat_emb[(c*100 + id)*16 + (e & 15)];
      else           vv = (float)xr[e - 90];        // numeric cols 6..37
      vals[j] = vv + tok_type[e];
    }
    float s = wredsum(vals[0]+vals[1]+vals[2]+vals[3]);
    const float mean = s * (1.0f/128.0f);
    float vsum = 0.0f;
    #pragma unroll
    for (int j = 0; j < 4; ++j){ float d = vals[j]-mean; vsum += d*d; }
    vsum = wredsum(vsum) * (1.0f/128.0f);
    const float inv = rsqrtf(vsum + 1e-12f);
    #pragma unroll
    for (int j = 0; j < 4; ++j){
      const int e = lane*4 + j;
      At[local*LDA + e] = (__bf16)((vals[j]-mean)*inv*lng[e] + lnb[e]);
    }
    // key-padding mask: sum of all 38 ints
    int ms = 0;
    if (lane < 19) ms = xr[2*lane] + xr[2*lane+1];
    ms = iredsum(ms);
    if (lane == 0) maskNeg[t] = (ms != 0) ? 0.0f : NEGBIG;
  }
  if (w == 0) __builtin_amdgcn_s_wait_tensorcnt(1);   // map_w landed in W0
  __syncthreads();

  v8f acc[8];

  // ---- GEMM 1: hs = emb @ map_w + map_b ----
  zero_acc(acc);
  gemm_strip(At, W0, m0, lane, acc);
  #pragma unroll
  for (int nt = 0; nt < 8; ++nt){
    const float bn = map_b[nt*16 + nlo];
    #pragma unroll
    for (int j = 0; j < 8; ++j) acc[nt][j] += bn;
  }
  __syncthreads();                     // everyone done reading emb(At) + W0
  if (w == 0){
    __builtin_amdgcn_s_wait_tensorcnt(0);             // wk landed in W1
    tdm_load_w(wimg + 2*WIMG_ELEMS, (unsigned int)WIMG_BYTES);  // wv -> W0
  }
  // store hs strip back into At (reuse), export last-token rows as f32
  #pragma unroll
  for (int nt = 0; nt < 8; ++nt){
    const int col = nt*16 + nlo;
    #pragma unroll
    for (int vr = 0; vr < 8; ++vr){
      const int row = m0 + vr + 8*g;
      const float hv = acc[nt][vr];
      At[row*LDA + col] = (__bf16)hv;
      const int t = t0 + row;
      if (t % S_ == S_-1) hs_last[(size_t)(t / S_)*128 + col] = hv;
    }
  }
  __syncthreads();

  // ---- GEMM 2: k = hs @ wk + bk  (W1) ----
  zero_acc(acc);
  gemm_strip(At, W1, m0, lane, acc);
  #pragma unroll
  for (int nt = 0; nt < 8; ++nt){
    const int col = nt*16 + nlo;
    const float bn = bk[col];
    #pragma unroll
    for (int vr = 0; vr < 8; ++vr){
      const int row = m0 + vr + 8*g;
      kout[(size_t)(t0 + row)*128 + col] = (__bf16)(acc[nt][vr] + bn);
    }
  }
  __syncthreads();
  if (w == 0) __builtin_amdgcn_s_wait_tensorcnt(0);   // wv landed in W0
  __syncthreads();

  // ---- GEMM 3: v = hs @ wv + bv  (W0) ----
  zero_acc(acc);
  gemm_strip(At, W0, m0, lane, acc);
  #pragma unroll
  for (int nt = 0; nt < 8; ++nt){
    const int col = nt*16 + nlo;
    const float bn = bv[col];
    #pragma unroll
    for (int vr = 0; vr < 8; ++vr){
      const int row = m0 + vr + 8*g;
      vout[(size_t)(t0 + row)*128 + col] = (__bf16)(acc[nt][vr] + bn);
    }
  }
}

// =====================================================================
// Kernel 1b: q = hs_last @ wq + bq   (M = 1024, WMMA; wq via TDM)
// =====================================================================
__global__ __launch_bounds__(256)
void k_qgemm(const float* __restrict__ hs_last, const __bf16* __restrict__ wq_img,
             const float* __restrict__ bq, __bf16* __restrict__ qout)
{
  extern __shared__ __bf16 smem[];
  __bf16* At = smem;
  __bf16* W0 = smem + WIMG_ELEMS;
  const int tid = threadIdx.x, lane = tid & 31, w = tid >> 5;
  const int r0 = blockIdx.x * 128;
  if (w == 0) tdm_load_w(wq_img, (unsigned int)WIMG_BYTES);
  for (int idx = tid; idx < 128*128; idx += 256){
    const int r = idx >> 7, c = idx & 127;
    At[r*LDA + c] = (__bf16)hs_last[(size_t)(r0 + r)*128 + c];
  }
  if (w == 0) __builtin_amdgcn_s_wait_tensorcnt(0);
  __syncthreads();
  v8f acc[8];
  zero_acc(acc);
  gemm_strip(At, W0, w*16, lane, acc);
  const int g = lane >> 4, nlo = lane & 15;
  #pragma unroll
  for (int nt = 0; nt < 8; ++nt){
    const int col = nt*16 + nlo;
    const float bn = bq[col];
    #pragma unroll
    for (int vr = 0; vr < 8; ++vr){
      const int row = w*16 + vr + 8*g;
      qout[(size_t)(r0 + row)*128 + col] = (__bf16)(acc[nt][vr] + bn);
    }
  }
}

// =====================================================================
// Kernel 2: single-query relative-position attention per (b,h).
// scores[r] = (q·k_r + q·de_r + k_r·de_r)/sqrt(DH) + mask ; ctx = softmax·v
// de_r = dist_emb[2S-2-r]
// =====================================================================
__global__ __launch_bounds__(128)
void k_attn(const __bf16* __restrict__ qbuf, const __bf16* __restrict__ kbuf,
            const __bf16* __restrict__ vbuf, const float* __restrict__ dist_emb,
            const float* __restrict__ maskNeg, float* __restrict__ ctxg)
{
  __shared__ float qs[32];
  __shared__ float des[S_*32];
  __shared__ float sc[S_];
  __shared__ float ctxp[4][32];
  const int tid = threadIdx.x;
  const int b = blockIdx.x >> 2;
  const int h = blockIdx.x & 3;

  if (tid < 32) qs[tid] = (float)qbuf[(size_t)b*128 + h*32 + tid];
  for (int idx = tid; idx < S_*32; idx += 128){
    const int r = idx >> 5, d = idx & 31;
    des[idx] = dist_emb[(2*S_ - 2 - r)*32 + d];
  }
  __syncthreads();

  for (int r = tid; r < S_; r += 128){
    const __bf16* kp = kbuf + ((size_t)b*S_ + r)*128 + h*32;
    float qk = 0.0f, qde = 0.0f, kde = 0.0f;
    #pragma unroll
    for (int d = 0; d < 32; ++d){
      const float kv = (float)kp[d];
      const float de = des[r*32 + d];
      qk  += qs[d] * kv;
      qde += qs[d] * de;
      kde += kv   * de;
    }
    sc[r] = (qk + qde + kde) * INV_SCALE + maskNeg[(size_t)b*S_ + r];
  }
  __syncthreads();

  float m = -3.4e38f;
  for (int r = 0; r < S_; ++r) m = fmaxf(m, sc[r]);
  float sum = 0.0f;
  for (int r = 0; r < S_; ++r) sum += __expf(sc[r] - m);

  const int w = tid >> 5, lane = tid & 31;
  float acc = 0.0f;
  for (int r = w; r < S_; r += 4){
    const float p = __expf(sc[r] - m);
    acc += p * (float)vbuf[((size_t)b*S_ + r)*128 + h*32 + lane];
  }
  ctxp[w][lane] = acc;
  __syncthreads();
  if (tid < 32){
    const float tot = ctxp[0][tid] + ctxp[1][tid] + ctxp[2][tid] + ctxp[3][tid];
    ctxg[(size_t)b*128 + h*32 + tid] = tot / sum;
  }
}

// =====================================================================
// Kernel 3: last-token tail. One wave per row (b).
// =====================================================================
__device__ inline float gelu_tanh(float z){
  return 0.5f * z * (1.0f + tanhf(0.7978845608028654f * (z + 0.044715f*z*z*z)));
}
__device__ inline float gelu_exact(float z){
  return 0.5f * z * (1.0f + erff(z * 0.7071067811865476f));
}

__global__ __launch_bounds__(256)
void k_tail(const float* __restrict__ hs_last, const float* __restrict__ ctxg,
            const float* __restrict__ attn_w, const float* __restrict__ attn_b,
            const float* __restrict__ aln_g, const float* __restrict__ aln_b,
            const float* __restrict__ ffn_w1, const float* __restrict__ ffn_b1,
            const float* __restrict__ ffn_w2, const float* __restrict__ ffn_b2,
            const float* __restrict__ fln_g, const float* __restrict__ fln_b,
            const int* __restrict__ prod, const int* __restrict__ day,
            const float* __restrict__ prod_emb, const float* __restrict__ day_emb,
            const float* __restrict__ h1_w, const float* __restrict__ h1_b,
            const float* __restrict__ ln1_g, const float* __restrict__ ln1_b,
            const float* __restrict__ h2_w, const float* __restrict__ h2_b,
            const float* __restrict__ ln2_g, const float* __restrict__ ln2_b,
            const float* __restrict__ h3_w, const float* __restrict__ h3_b,
            float* __restrict__ out)
{
  const int lane = threadIdx.x & 31;
  const int b = blockIdx.x * 8 + (threadIdx.x >> 5);

  float hsr[4], cx[4];
  #pragma unroll
  for (int j = 0; j < 4; ++j){
    const int e = lane*4 + j;
    hsr[j] = hs_last[(size_t)b*128 + e];
    cx[j]  = ctxg[(size_t)b*128 + e];
  }

  // ao_pre = hs + ctx @ attn_w + attn_b
  float ao[4];
  #pragma unroll
  for (int j = 0; j < 4; ++j) ao[j] = attn_b[lane*4 + j] + hsr[j];
  for (int d0 = 0; d0 < 128; d0 += 4){
    const int src = d0 >> 2;
    const float c0 = __shfl(cx[0], src), c1 = __shfl(cx[1], src);
    const float c2 = __shfl(cx[2], src), c3 = __shfl(cx[3], src);
    #pragma unroll
    for (int j = 0; j < 4; ++j){
      const int e = lane*4 + j;
      ao[j] += c0*attn_w[(d0+0)*128+e] + c1*attn_w[(d0+1)*128+e]
             + c2*attn_w[(d0+2)*128+e] + c3*attn_w[(d0+3)*128+e];
    }
  }
  {
    float s = wredsum(ao[0]+ao[1]+ao[2]+ao[3]) * (1.0f/128.0f);
    float v = 0.0f;
    #pragma unroll
    for (int j = 0; j < 4; ++j){ float d = ao[j]-s; v += d*d; }
    v = wredsum(v) * (1.0f/128.0f);
    const float inv = rsqrtf(v + 1e-12f);
    #pragma unroll
    for (int j = 0; j < 4; ++j){
      const int e = lane*4 + j;
      ao[j] = (ao[j]-s)*inv*aln_g[e] + aln_b[e];
    }
  }
  // ffn = gelu_tanh(ao @ w1 + b1)
  float f[4];
  #pragma unroll
  for (int j = 0; j < 4; ++j) f[j] = ffn_b1[lane*4 + j];
  for (int d0 = 0; d0 < 128; d0 += 4){
    const int src = d0 >> 2;
    const float a0 = __shfl(ao[0], src), a1 = __shfl(ao[1], src);
    const float a2 = __shfl(ao[2], src), a3 = __shfl(ao[3], src);
    #pragma unroll
    for (int j = 0; j < 4; ++j){
      const int e = lane*4 + j;
      f[j] += a0*ffn_w1[(d0+0)*128+e] + a1*ffn_w1[(d0+1)*128+e]
            + a2*ffn_w1[(d0+2)*128+e] + a3*ffn_w1[(d0+3)*128+e];
    }
  }
  #pragma unroll
  for (int j = 0; j < 4; ++j) f[j] = gelu_tanh(f[j]);
  // seq = LN(ao + f @ w2 + b2)
  float sq[4];
  #pragma unroll
  for (int j = 0; j < 4; ++j) sq[j] = ffn_b2[lane*4 + j] + ao[j];
  for (int d0 = 0; d0 < 128; d0 += 4){
    const int src = d0 >> 2;
    const float a0 = __shfl(f[0], src), a1 = __shfl(f[1], src);
    const float a2 = __shfl(f[2], src), a3 = __shfl(f[3], src);
    #pragma unroll
    for (int j = 0; j < 4; ++j){
      const int e = lane*4 + j;
      sq[j] += a0*ffn_w2[(d0+0)*128+e] + a1*ffn_w2[(d0+1)*128+e]
             + a2*ffn_w2[(d0+2)*128+e] + a3*ffn_w2[(d0+3)*128+e];
    }
  }
  {
    float s = wredsum(sq[0]+sq[1]+sq[2]+sq[3]) * (1.0f/128.0f);
    float v = 0.0f;
    #pragma unroll
    for (int j = 0; j < 4; ++j){ float d = sq[j]-s; v += d*d; }
    v = wredsum(v) * (1.0f/128.0f);
    const float inv = rsqrtf(v + 1e-12f);
    #pragma unroll
    for (int j = 0; j < 4; ++j){
      const int e = lane*4 + j;
      sq[j] = (sq[j]-s)*inv*fln_g[e] + fln_b[e];
    }
  }
  // head: feat = [sq(128), prod_emb(16), day_emb(16)]; out col t = lane
  const int pb = prod[b], db = day[b];
  float z = h1_b[lane];
  for (int i0 = 0; i0 < 128; i0 += 4){
    const int src = i0 >> 2;
    const float a0 = __shfl(sq[0], src), a1 = __shfl(sq[1], src);
    const float a2 = __shfl(sq[2], src), a3 = __shfl(sq[3], src);
    z += a0*h1_w[(i0+0)*32+lane] + a1*h1_w[(i0+1)*32+lane]
       + a2*h1_w[(i0+2)*32+lane] + a3*h1_w[(i0+3)*32+lane];
  }
  #pragma unroll
  for (int i = 0; i < 16; ++i) z += prod_emb[pb*16 + i] * h1_w[(128+i)*32 + lane];
  #pragma unroll
  for (int i = 0; i < 16; ++i) z += day_emb[db*16 + i] * h1_w[(144+i)*32 + lane];
  {
    const float m = wredsum(z) * (1.0f/32.0f);
    float d = z - m;
    const float v = wredsum(d*d) * (1.0f/32.0f);
    z = gelu_exact(d * rsqrtf(v + 1e-5f) * ln1_g[lane] + ln1_b[lane]);
  }
  float z2 = h2_b[lane];
  for (int i = 0; i < 32; ++i) z2 += __shfl(z, i) * h2_w[i*32 + lane];
  {
    const float m = wredsum(z2) * (1.0f/32.0f);
    float d = z2 - m;
    const float v = wredsum(d*d) * (1.0f/32.0f);
    z2 = gelu_exact(d * rsqrtf(v + 1e-5f) * ln2_g[lane] + ln2_b[lane]);
  }
  float o = wredsum(z2 * h3_w[lane]);
  if (lane == 0) out[b] = o + h3_b[0];
}

// =====================================================================
extern "C" void kernel_launch(void* const* d_in, const int* in_sizes, int n_in,
                              void* d_out, int out_size, void* d_ws, size_t ws_size,
                              hipStream_t stream) {
  (void)in_sizes; (void)n_in; (void)out_size; (void)ws_size;
  const int*   x        = (const int*)  d_in[0];
  const int*   prod     = (const int*)  d_in[1];
  const int*   day      = (const int*)  d_in[2];
  const float* cat_emb  = (const float*)d_in[3];
  const float* prod_emb = (const float*)d_in[4];
  const float* day_emb  = (const float*)d_in[5];
  const float* tok_type = (const float*)d_in[6];
  const float* emb_ln_g = (const float*)d_in[7];
  const float* emb_ln_b = (const float*)d_in[8];
  const float* map_w    = (const float*)d_in[9];
  const float* map_b    = (const float*)d_in[10];
  const float* wq       = (const float*)d_in[11];
  const float* bq       = (const float*)d_in[12];
  const float* wk       = (const float*)d_in[13];
  const float* bk       = (const float*)d_in[14];
  const float* wv       = (const float*)d_in[15];
  const float* bv       = (const float*)d_in[16];
  const float* attn_w   = (const float*)d_in[17];
  const float* attn_b   = (const float*)d_in[18];
  const float* dist_emb = (const float*)d_in[19];
  const float* aln_g    = (const float*)d_in[20];
  const float* aln_b    = (const float*)d_in[21];
  const float* ffn_w1   = (const float*)d_in[22];
  const float* ffn_b1   = (const float*)d_in[23];
  const float* ffn_w2   = (const float*)d_in[24];
  const float* ffn_b2   = (const float*)d_in[25];
  const float* fln_g    = (const float*)d_in[26];
  const float* fln_b    = (const float*)d_in[27];
  const float* h1_w     = (const float*)d_in[28];
  const float* h1_b     = (const float*)d_in[29];
  const float* ln1_g    = (const float*)d_in[30];
  const float* ln1_b    = (const float*)d_in[31];
  const float* h2_w     = (const float*)d_in[32];
  const float* h2_b     = (const float*)d_in[33];
  const float* ln2_g    = (const float*)d_in[34];
  const float* ln2_b    = (const float*)d_in[35];
  const float* h3_w     = (const float*)d_in[36];
  const float* h3_b     = (const float*)d_in[37];
  float* out = (float*)d_out;

  char* ws = (char*)d_ws;
  __bf16* kbuf    = (__bf16*)(ws);                                  // NTOK*128 bf16
  __bf16* vbuf    = (__bf16*)(ws + 36438016);                       // NTOK*128 bf16
  float*  hs_last = (float*) (ws + 72876032);                       // B*128 f32
  __bf16* qbuf    = (__bf16*)(ws + 73400320);                       // B*128 bf16
  float*  ctxg    = (float*) (ws + 73662464);                       // B*128 f32
  float*  maskNeg = (float*) (ws + 74186752);                       // NTOK f32
  __bf16* wimgs   = (__bf16*)(ws + 74756096);                       // 4 x 33280B images

  k_prep_w<<<4, 256, 0, stream>>>(map_w, wk, wv, wq, wimgs);

  k_embed_map_kv<<<NTOK/128, 256, 3*WIMG_BYTES, stream>>>(
      x, cat_emb, tok_type, emb_ln_g, emb_ln_b,
      wimgs, map_b, bk, bv, kbuf, vbuf, hs_last, maskNeg);

  k_qgemm<<<B_/128, 256, 2*WIMG_BYTES, stream>>>(
      hs_last, wimgs + 3*WIMG_ELEMS, bq, qbuf);

  k_attn<<<B_*4, 128, 0, stream>>>(qbuf, kbuf, vbuf, dist_emb, maskNeg, ctxg);

  k_tail<<<B_/8, 256, 0, stream>>>(hs_last, ctxg, attn_w, attn_b, aln_g, aln_b,
      ffn_w1, ffn_b1, ffn_w2, ffn_b2, fln_g, fln_b,
      prod, day, prod_emb, day_emb,
      h1_w, h1_b, ln1_g, ln1_b, h2_w, h2_b, ln2_g, ln2_b, h3_w, h3_b, out);
}
